// SelfAttention_26379689132825
// MI455X (gfx1250) — matrix operand order repaired and checked by reading, not compiled, not run on hardware
//
#include <hip/hip_runtime.h>

// ---------------------------------------------------------------------------
// Self-attention, B=4 S=4096 D=512, fp32 in/out, bf16 WMMA compute (gfx1250).
// Pipeline (all operands pre-swizzled into WMMA fragment order in global):
//   prep_w x4      : W fp32 -> bf16 B-frag order
//   gemm_proj x3   : proj = X@W+b, outputs written in consumer frag order
//   flash_attn     : Q via async global->LDS copy, K/V frags direct from L2,
//                    online softmax, output via async LDS->global store
//   gemm_out       : A/B frags direct from global, fp32 row-major + bias
// Workspace: qp/kp/vp/ao (16 MiB each, bf16) + 4 weight frag buffers (~66 MiB)
// ---------------------------------------------------------------------------

typedef __attribute__((ext_vector_type(16))) __bf16 v16bf;
typedef __attribute__((ext_vector_type(8)))  float  v8f;

static constexpr int EMB   = 512;
static constexpr int SEQ   = 4096;
static constexpr int BATCH = 4;
static constexpr size_t MD = (size_t)BATCH * SEQ * EMB;   // 8388608 elems
static constexpr size_t WSZ = (size_t)EMB * EMB;          // 262144 elems

// Fragment layouts (wave32, 16-bit operands, CDNA5 ISA):
//  A (16x32):  lane = (r&15) + 16*((k>>3)&1) ; elem = (k&7) + 8*((k&31)>>4)
//              -> 8 consecutive k of one row are contiguous in one lane.
//  B (32x16):  lane = (n&15) + 16*((k>>3)&1) ; elem = (k&7) + 8*((k&31)>>4)
//  C/D (16x16 f32): vgpr v -> M = v + 8*(lane>>4), N = lane&15
// A fragment = 512 x bf16 = 1 KiB; per lane 16 contiguous bf16 (32 B).

__device__ __forceinline__ unsigned short f2bf(float x) {
    union { float f; unsigned u; } c; c.f = x;
    unsigned u = c.u;
    u += 0x7fffu + ((u >> 16) & 1u);          // round-to-nearest-even
    return (unsigned short)(u >> 16);
}

__device__ __forceinline__ v16bf ldfragL(const unsigned short* p, int frag, int lane) {
    return *reinterpret_cast<const v16bf*>(p + frag * 512 + lane * 16);   // LDS
}
__device__ __forceinline__ v16bf ldfragG(const unsigned short* p, size_t frag, int lane) {
    return *reinterpret_cast<const v16bf*>(p + frag * 512 + lane * 16);   // global
}

__device__ __forceinline__ v8f wmma_bf16(v16bf a, v16bf b, v8f c) {
    return __builtin_amdgcn_wmma_f32_16x16x32_bf16(false, a, false, b,
                                                   (short)0, c, false, false);
}

__device__ __forceinline__ unsigned lds_off(const void* p) {
    return (unsigned)(uintptr_t)p;            // low 32 bits = LDS byte offset
}
// CDNA5 async copy: global -> LDS, 16B per lane, tracked by ASYNCcnt.
__device__ __forceinline__ void async_g2l_b128(unsigned lds_addr, unsigned goff,
                                               const void* sbase) {
    asm volatile("global_load_async_to_lds_b128 %0, %1, %2"
                 :: "v"(lds_addr), "v"(goff), "s"(sbase) : "memory");
}
__device__ __forceinline__ void async_l2g_b128(unsigned goff, unsigned lds_addr,
                                               const void* sbase) {
    asm volatile("global_store_async_from_lds_b128 %0, %1, %2"
                 :: "v"(goff), "v"(lds_addr), "s"(sbase) : "memory");
}
__device__ __forceinline__ void wait_async0() {
    asm volatile("s_wait_asynccnt 0x0" ::: "memory");
}

// ---------------------------------------------------------------------------
// prep_w: W[512,512] fp32 row-major -> bf16 B-fragment order [kf16][nf32][512]
// ---------------------------------------------------------------------------
__global__ __launch_bounds__(256)
void prep_w_kernel(const float* __restrict__ W, unsigned short* __restrict__ wb) {
    int t = blockIdx.x * 256 + threadIdx.x;   // 16384 threads: two 16B chunks each
    int n = t & 511;                          // output column
    int h = t >> 9;                           // k block of 16 (0..31)
    unsigned short tmp[16];
#pragma unroll
    for (int i = 0; i < 16; ++i)
        tmp[i] = f2bf(W[(size_t)(h * 16 + i) * EMB + n]);   // coalesced over n
    int frag = (h >> 1) * 32 + (n >> 4);
    int lt = (n & 15);
    uint4* dst = (uint4*)(wb + (size_t)frag * 512 + lt * 16 + 8 * (h & 1));
    const uint4* src = (const uint4*)tmp;
    dst[0] = src[0]; dst[32] = src[1];
}

// ---------------------------------------------------------------------------
// gemm_proj: Y = X(fp32)@W + b, tile 64x64, 4 waves. B-frags direct from
// global; A staged via LDS with 16B-chunk permutation; epilogue bounces the
// result into the consumer's fragment order.
//   OUT_QA: A-frag  [m/16][d/32]            (flash Q / P side)
//   OUT_KB: B-frag  [b][key/16][d/32]       (flash QK^T B operand)
//   OUT_VB: B-frag  [b][key/32][d/16]       (flash PV   B operand)
// ---------------------------------------------------------------------------
enum { OUT_QA = 0, OUT_KB = 1, OUT_VB = 2 };

template <int LAYOUT>
__global__ __launch_bounds__(128)
void gemm_proj_kernel(const float* __restrict__ X, const unsigned short* __restrict__ wb,
                      const float* __restrict__ bias, unsigned short* __restrict__ Y) {
    __shared__ __align__(32) unsigned short la[4 * 512];   // 4 A-frags (64x32)
    __shared__ __align__(32) unsigned short lo[8 * 512];   // 64x64 out bounce

    const int tid  = threadIdx.x;
    const int lane = tid & 31;
    const int wave = tid >> 5;
    const int wm = wave >> 1, wn = wave & 1;
    const int m0 = blockIdx.x * 64;
    const int n0 = blockIdx.y * 64;

    v8f acc[2][2] = {};
    const int arow = (tid * 16) >> 5;         // staged A row for this thread
    const int gbase = ((tid * 16) & 31) >> 3; // first 8-elem group (0 or 2)

    for (int kf = 0; kf < EMB / 32; ++kf) {
        const int k0 = kf * 32;
        // ---- stage A tile 64x32: two 16B-chunk permuted stores per thread
#pragma unroll
        for (int g2 = 0; g2 < 2; ++g2) {
            int g = gbase + g2;               // 8-elem group within the 32-k tile
            const float* src = X + (size_t)(m0 + arow) * EMB + k0 + g * 8;
            float4 f0 = *(const float4*)(src);
            float4 f1 = *(const float4*)(src + 4);
            unsigned short tmp[8] = {f2bf(f0.x), f2bf(f0.y), f2bf(f0.z), f2bf(f0.w),
                                     f2bf(f1.x), f2bf(f1.y), f2bf(f1.z), f2bf(f1.w)};
            int lt = (arow & 15) + 16 * (g & 1);
            *(uint4*)(la + (arow >> 4) * 512 + lt * 16 + 8 * (g >> 1)) = *(const uint4*)tmp;
        }
        __syncthreads();
        v16bf a0 = ldfragL(la, wm * 2 + 0, lane);
        v16bf a1 = ldfragL(la, wm * 2 + 1, lane);
        v16bf b0 = ldfragG(wb, (size_t)kf * 32 + (n0 >> 4) + wn * 2 + 0, lane);
        v16bf b1 = ldfragG(wb, (size_t)kf * 32 + (n0 >> 4) + wn * 2 + 1, lane);
        acc[0][0] = wmma_bf16(a0, b0, acc[0][0]);
        acc[0][1] = wmma_bf16(a0, b1, acc[0][1]);
        acc[1][0] = wmma_bf16(a1, b0, acc[1][0]);
        acc[1][1] = wmma_bf16(a1, b1, acc[1][1]);
        __syncthreads();
    }

    // ---- epilogue: bias, bounce to target fragment layout in LDS
#pragma unroll
    for (int mi = 0; mi < 2; ++mi)
#pragma unroll
        for (int nj = 0; nj < 2; ++nj)
#pragma unroll
            for (int v = 0; v < 8; ++v) {
                int rl = wm * 32 + mi * 16 + v + 8 * (lane >> 4);   // tile row
                int cl = wn * 32 + nj * 16 + (lane & 15);           // tile col (=d)
                unsigned short bv = f2bf(acc[mi][nj][v] + bias[n0 + cl]);
                int off;
                if constexpr (LAYOUT == OUT_QA) {
                    int cc = cl & 31;
                    off = ((rl >> 4) * 2 + (cl >> 5)) * 512 +
                          ((rl & 15) + 16 * ((cc >> 3) & 1)) * 16 +
                          (cc & 7) + 8 * (cc >> 4);
                } else if constexpr (LAYOUT == OUT_KB) {
                    off = ((rl >> 4) * 2 + (cl >> 5)) * 512 +
                          ((rl & 15) + 16 * ((cl >> 3) & 1)) * 16 + (cl & 7) + 8 * ((cl >> 4) & 1);
                } else {  // OUT_VB
                    off = ((rl >> 5) * 4 + (cl >> 4)) * 512 +
                          ((cl & 15) + 16 * ((rl >> 3) & 1)) * 16 + (rl & 7) + 8 * ((rl >> 4) & 1);
                }
                lo[off] = bv;
            }
    __syncthreads();
    // ---- copy 8 local frags to their global frag slots (16B chunks)
    const int batch   = m0 >> 12;             // 4096 rows per batch
    const int rInB    = m0 & 4095;
#pragma unroll
    for (int i = 0; i < 4; ++i) {
        int c  = tid + 128 * i;               // 512 chunks of 16B
        int lf = c >> 6;
        size_t gf;
        if constexpr (LAYOUT == OUT_QA) {
            gf = (size_t)((m0 >> 4) + (lf >> 1)) * 16 + (n0 >> 5) + (lf & 1);
        } else if constexpr (LAYOUT == OUT_KB) {
            gf = ((size_t)batch * 256 + (rInB >> 4) + (lf >> 1)) * 16 + (n0 >> 5) + (lf & 1);
        } else {  // OUT_VB
            gf = ((size_t)batch * 128 + (rInB >> 5) + (lf >> 2)) * 32 + (n0 >> 4) + (lf & 3);
        }
        *(uint4*)(Y + gf * 512 + (size_t)(c & 63) * 8) = *(const uint4*)(lo + lf * 512 + (c & 63) * 8);
    }
}

// ---------------------------------------------------------------------------
// Flash attention: 32 query rows x D=512 per block, key tiles of 128.
// Q: async global->LDS copy (contiguous frag block). K/V: fragments read
// directly from global (L2-resident). Static LDS ~57 KB.
// ---------------------------------------------------------------------------
__global__ __launch_bounds__(256)
void flash_attn_kernel(const unsigned short* __restrict__ qp,
                       const unsigned short* __restrict__ kp,
                       const unsigned short* __restrict__ vp,
                       unsigned short* __restrict__ ao) {
    __shared__ __align__(32) unsigned short qf[32 * 512];  // Q A-frags / out bounce
    __shared__ __align__(32) unsigned short pf[8 * 512];   // P A-frags (32x128)
    __shared__ float sS[32 * 128];
    __shared__ float red[32 * 8];
    __shared__ float mrow[32], lrow[32], scl[32];

    const int tid = threadIdx.x, lane = tid & 31, wave = tid >> 5;
    const int batch = blockIdx.x / (SEQ / 32);
    const int qt    = blockIdx.x % (SEQ / 32);
    const int qrow0 = batch * SEQ + qt * 32;

    // ---- async copy Q tile (32 contiguous frags = 32 KiB) into LDS
    {
        const unsigned qlds = lds_off(qf);
        const unsigned gb   = (unsigned)((size_t)qrow0 * EMB * 2);  // byte offset
#pragma unroll
        for (int i = 0; i < 8; ++i) {
            unsigned c = (unsigned)tid + 256u * i;                  // 16B chunk id
            async_g2l_b128(qlds + c * 16u, gb + c * 16u, (const void*)qp);
        }
    }
    if (tid < 32) { mrow[tid] = -3.0e38f; lrow[tid] = 0.f; }
    v8f o[2][4] = {};
    wait_async0();
    __syncthreads();

    const int srw = tid >> 3;   // softmax row for this thread
    const int spt = tid & 7;    // 16-col slice

    for (int kt = 0; kt < SEQ / 128; ++kt) {
        // ---- S strip = Q @ K^T (this wave: 32 rows x 16 keys), K frags from L2
        const unsigned short* kpB =
            kp + ((size_t)batch * 256 + (size_t)kt * 8 + wave) * 16 * 512;
        v8f sa[2] = {};
#pragma unroll
        for (int df = 0; df < 16; ++df) {
            v16bf b  = ldfragG(kpB, df, lane);
            v16bf a0 = ldfragL(qf, 0 * 16 + df, lane);
            v16bf a1 = ldfragL(qf, 1 * 16 + df, lane);
            sa[0] = wmma_bf16(a0, b, sa[0]);
            sa[1] = wmma_bf16(a1, b, sa[1]);
        }
        const float isq = 0.04419417382415922f;   // 1/sqrt(512)
#pragma unroll
        for (int mi = 0; mi < 2; ++mi)
#pragma unroll
            for (int v = 0; v < 8; ++v)
                sS[(mi * 16 + v + 8 * (lane >> 4)) * 128 + wave * 16 + (lane & 15)] =
                    sa[mi][v] * isq;
        __syncthreads();
        // ---- online softmax
        float lm = -3.0e38f;
#pragma unroll
        for (int c = 0; c < 16; ++c) lm = fmaxf(lm, sS[srw * 128 + spt * 16 + c]);
        red[srw * 8 + spt] = lm;
        __syncthreads();
        if (spt == 0) {
            float mo = mrow[srw], mx = mo;
#pragma unroll
            for (int p = 0; p < 8; ++p) mx = fmaxf(mx, red[srw * 8 + p]);
            float s = __expf(mo - mx);
            mrow[srw] = mx; lrow[srw] *= s; scl[srw] = s;
        }
        __syncthreads();
        {   // P = exp(S-m) into bf16 A-frags + partial sums
            float mx = mrow[srw], ls = 0.f;
#pragma unroll
            for (int c = 0; c < 16; ++c) {
                int col = spt * 16 + c;
                float p = __expf(sS[srw * 128 + col] - mx);
                ls += p;
                int kk = col & 31;
                pf[((srw >> 4) * 4 + (col >> 5)) * 512 +
                   ((srw & 15) + 16 * ((kk >> 3) & 1)) * 16 +
                   (kk & 7) + 8 * (kk >> 4)] = f2bf(p);
            }
            red[srw * 8 + spt] = ls;
        }
        // ---- rescale running O
#pragma unroll
        for (int mi = 0; mi < 2; ++mi)
#pragma unroll
            for (int v = 0; v < 8; ++v) {
                float s = scl[mi * 16 + v + 8 * (lane >> 4)];
#pragma unroll
                for (int nj = 0; nj < 4; ++nj) o[mi][nj][v] *= s;
            }
        __syncthreads();
        if (spt == 0) {
            float ss = 0.f;
#pragma unroll
            for (int p = 0; p < 8; ++p) ss += red[srw * 8 + p];
            lrow[srw] += ss;
        }
        // ---- O += P @ V ; V frags direct from L2 (this wave's 64-wide d-chunk)
#pragma unroll
        for (int kf = 0; kf < 4; ++kf) {
            v16bf a0 = ldfragL(pf, 0 * 4 + kf, lane);
            v16bf a1 = ldfragL(pf, 1 * 4 + kf, lane);
            const unsigned short* vpB =
                vp + (((size_t)batch * 128 + (size_t)kt * 4 + kf) * 32 + wave * 4) * 512;
#pragma unroll
            for (int nj = 0; nj < 4; ++nj) {
                v16bf b = ldfragG(vpB, nj, lane);
                o[0][nj] = wmma_bf16(a0, b, o[0][nj]);
                o[1][nj] = wmma_bf16(a1, b, o[1][nj]);
            }
        }
        __syncthreads();
    }
    // ---- normalize, bounce into A-frag order (reusing qf), async store out
    __syncthreads();
#pragma unroll
    for (int mi = 0; mi < 2; ++mi)
#pragma unroll
        for (int nj = 0; nj < 4; ++nj)
#pragma unroll
            for (int v = 0; v < 8; ++v) {
                int rl = mi * 16 + v + 8 * (lane >> 4);
                int d  = wave * 64 + nj * 16 + (lane & 15);
                float val = o[mi][nj][v] / lrow[rl];
                int dd = d & 31;
                qf[((rl >> 4) * 16 + (d >> 5)) * 512 +
                   ((rl & 15) + 16 * ((dd >> 3) & 1)) * 16 +
                   (dd & 7) + 8 * (dd >> 4)] = f2bf(val);
            }
    __syncthreads();
    {
        const unsigned qlds = lds_off(qf);
        const unsigned gb   = (unsigned)((size_t)qrow0 * EMB * 2);
#pragma unroll
        for (int i = 0; i < 8; ++i) {
            unsigned c = (unsigned)tid + 256u * i;
            async_l2g_b128(gb + c * 16u, qlds + c * 16u, (const void*)ao);
        }
        wait_async0();
    }
}

// ---------------------------------------------------------------------------
// gemm_out: A-frags (ao) and B-frags (Wo) direct from global; fp32 out + bias
// ---------------------------------------------------------------------------
__global__ __launch_bounds__(128)
void gemm_out_kernel(const unsigned short* __restrict__ af,
                     const unsigned short* __restrict__ wb,
                     const float* __restrict__ bias, float* __restrict__ out) {
    const int tid = threadIdx.x, lane = tid & 31, wave = tid >> 5;
    const int wm = wave >> 1, wn = wave & 1;
    const int m0 = blockIdx.x * 64;
    const int n0 = blockIdx.y * 64;

    v8f acc[2][2] = {};
#pragma unroll 4
    for (int kf = 0; kf < EMB / 32; ++kf) {
        v16bf a0 = ldfragG(af, (size_t)((m0 >> 4) + wm * 2 + 0) * 16 + kf, lane);
        v16bf a1 = ldfragG(af, (size_t)((m0 >> 4) + wm * 2 + 1) * 16 + kf, lane);
        v16bf b0 = ldfragG(wb, (size_t)kf * 32 + (n0 >> 4) + wn * 2 + 0, lane);
        v16bf b1 = ldfragG(wb, (size_t)kf * 32 + (n0 >> 4) + wn * 2 + 1, lane);
        acc[0][0] = wmma_bf16(a0, b0, acc[0][0]);
        acc[0][1] = wmma_bf16(a0, b1, acc[0][1]);
        acc[1][0] = wmma_bf16(a1, b0, acc[1][0]);
        acc[1][1] = wmma_bf16(a1, b1, acc[1][1]);
    }
#pragma unroll
    for (int mi = 0; mi < 2; ++mi)
#pragma unroll
        for (int nj = 0; nj < 2; ++nj)
#pragma unroll
            for (int v = 0; v < 8; ++v) {
                int row = m0 + wm * 32 + mi * 16 + v + 8 * (lane >> 4);
                int col = n0 + wn * 32 + nj * 16 + (lane & 15);
                out[(size_t)row * EMB + col] = acc[mi][nj][v] + bias[col];
            }
}

// ---------------------------------------------------------------------------
extern "C" void kernel_launch(void* const* d_in, const int* in_sizes, int n_in,
                              void* d_out, int out_size, void* d_ws, size_t ws_size,
                              hipStream_t stream) {
    (void)in_sizes; (void)n_in; (void)out_size; (void)ws_size;
    const float* k  = (const float*)d_in[0];
    const float* v  = (const float*)d_in[1];
    const float* q  = (const float*)d_in[2];
    const float* Wk = (const float*)d_in[3];
    const float* bk = (const float*)d_in[4];
    const float* Wv = (const float*)d_in[5];
    const float* bv = (const float*)d_in[6];
    const float* Wq = (const float*)d_in[7];
    const float* bq = (const float*)d_in[8];
    const float* Wo = (const float*)d_in[9];
    const float* bo = (const float*)d_in[10];

    unsigned short* qp  = (unsigned short*)d_ws;
    unsigned short* kp  = qp + MD;
    unsigned short* vp  = kp + MD;
    unsigned short* ao  = vp + MD;
    unsigned short* wbq = ao + MD;
    unsigned short* wbk = wbq + WSZ;
    unsigned short* wbv = wbk + WSZ;
    unsigned short* wbo = wbv + WSZ;

    prep_w_kernel<<<64, 256, 0, stream>>>(Wq, wbq);
    prep_w_kernel<<<64, 256, 0, stream>>>(Wk, wbk);
    prep_w_kernel<<<64, 256, 0, stream>>>(Wv, wbv);
    prep_w_kernel<<<64, 256, 0, stream>>>(Wo, wbo);

    dim3 gGrid(BATCH * SEQ / 64, EMB / 64);
    gemm_proj_kernel<OUT_QA><<<gGrid, 128, 0, stream>>>(q, wbq, bq, qp);
    gemm_proj_kernel<OUT_KB><<<gGrid, 128, 0, stream>>>(k, wbk, bk, kp);
    gemm_proj_kernel<OUT_VB><<<gGrid, 128, 0, stream>>>(v, wbv, bv, vp);

    flash_attn_kernel<<<dim3(BATCH * SEQ / 32), 256, 0, stream>>>(qp, kp, vp, ao);

    gemm_out_kernel<<<gGrid, 128, 0, stream>>>(ao, wbo, bo, (float*)d_out);
}
